// QuantLinear_64733747085670
// MI455X (gfx1250) — matrix operand order repaired and checked by reading, not compiled, not run on hardware
//
#include <hip/hip_runtime.h>

typedef __attribute__((ext_vector_type(16))) _Float16 v16h;
typedef __attribute__((ext_vector_type(8)))  _Float16 v8h;
typedef __attribute__((ext_vector_type(4)))  _Float16 v4h;
typedef __attribute__((ext_vector_type(8)))  float    v8f;

#define BM 128
#define BN 128
#define BK 64
#define LDA 72   // As row stride in halves (144 B, 16B-aligned, padded vs 64)
#define LDB 72   // Bs row stride in halves
#define NTHREADS 256

// D = X * W, W[k,n] = scale[g,n] * (q4[k,n] - (z4[g,n]+1)), g = k/128
// f16 WMMA (16x16x32), f32 accumulate, double-buffered LDS pipeline.
__global__ __launch_bounds__(NTHREADS)
void gptq_w4a16_wmma(const float* __restrict__ X,
                     const int*   __restrict__ QW,
                     const int*   __restrict__ QZ,
                     const float* __restrict__ SC,
                     float* __restrict__ Out,
                     int M, int N, int K)
{
    __shared__ _Float16 As[2][BM * LDA];   // [m][k] f16
    __shared__ _Float16 Bs[2][BN * LDB];   // [n][k] f16 (transposed)

    const int tid   = threadIdx.x;
    const int lane  = tid & 31;
    const int wave  = tid >> 5;
    const int waveM = wave >> 1;        // 0..3
    const int waveN = wave & 1;         // 0..1

    const int bn0 = blockIdx.x * BN;
    const int bm0 = blockIdx.y * BM;

    const int nPack   = N >> 3;
    const int kChunks = K / BK;

    // dequant ownership: one column per thread, two packed-row parities
    const int wn   = tid & (BN - 1);    // 0..127
    const int prb  = tid >> 7;          // 0 or 1
    const int ncol = bn0 + wn;

    // X staging ownership
    const int xr0 = tid >> 4;           // base row parity (0..15)
    const int xc4 = tid & 15;           // float4 column within chunk

    v8f acc[2][4] = {};

    // WMMA f16 operand lane mapping (ISA 7.12.2)
    const int aoff = (lane & 16) ? 8 : 0;    // A: K sub-offset per half-wave
    const int boff = (lane & 16) ? 8 : 0;    // B: K sub-offset per half-wave
    const int l15  = lane & 15;

    float4 xr[8];
    int    wr[4];
    float  scale, mzs;

    // ---- prologue: fetch + stage chunk 0 into buffer 0 ----
    {
        #pragma unroll
        for (int i = 0; i < 8; ++i)
            xr[i] = *reinterpret_cast<const float4*>(
                X + (size_t)(bm0 + i * 16 + xr0) * K + xc4 * 4);
        scale = SC[ncol];
        const int zp = ((QZ[ncol >> 3] >> ((ncol & 7) * 4)) & 0xF) + 1;
        mzs = -scale * (float)zp;
        #pragma unroll
        for (int i = 0; i < 4; ++i)
            wr[i] = QW[(size_t)(prb + 2 * i) * N + ncol];

        #pragma unroll
        for (int i = 0; i < 8; ++i) {
            v4h h = { (_Float16)xr[i].x, (_Float16)xr[i].y,
                      (_Float16)xr[i].z, (_Float16)xr[i].w };
            *reinterpret_cast<v4h*>(&As[0][(i * 16 + xr0) * LDA + xc4 * 4]) = h;
        }
        #pragma unroll
        for (int i = 0; i < 4; ++i) {
            const int pr = prb + 2 * i;
            v8h wq;
            #pragma unroll
            for (int j = 0; j < 8; ++j)
                wq[j] = (_Float16)(scale * (float)((wr[i] >> (4 * j)) & 0xF) + mzs);
            *reinterpret_cast<v8h*>(&Bs[0][wn * LDB + pr * 8]) = wq;
        }
    }
    __syncthreads();

    for (int kc = 0; kc < kChunks; ++kc) {
        const int  cur     = kc & 1;
        const bool hasNext = (kc + 1 < kChunks);

        // ---- issue next chunk's global loads (consumed after compute) ----
        if (hasNext) {
            const int kb2 = (kc + 1) * BK;
            #pragma unroll
            for (int i = 0; i < 8; ++i)
                xr[i] = *reinterpret_cast<const float4*>(
                    X + (size_t)(bm0 + i * 16 + xr0) * K + kb2 + xc4 * 4);
            const int g = kb2 >> 7;  // GROUP_SIZE == 128
            scale = SC[(size_t)g * N + ncol];
            const int zp = ((QZ[(size_t)g * nPack + (ncol >> 3)]
                             >> ((ncol & 7) * 4)) & 0xF) + 1;
            mzs = -scale * (float)zp;
            #pragma unroll
            for (int i = 0; i < 4; ++i)
                wr[i] = QW[(size_t)((kb2 >> 3) + prb + 2 * i) * N + ncol];
            if (kc + 2 < kChunks) {
                __builtin_prefetch(X + (size_t)(bm0 + (tid >> 1)) * K
                                     + kb2 + BK + (tid & 1) * 32, 0, 1);
                __builtin_prefetch(QW + (size_t)(((kb2 + BK) >> 3) + prb) * N + ncol, 0, 1);
            }
        }

        // ---- compute current chunk: 2 k-steps x (2 M x 4 N) WMMAs ----
        const _Float16* Ab = As[cur];
        const _Float16* Bb = Bs[cur];
        #pragma unroll
        for (int kk = 0; kk < BK; kk += 32) {
            v16h afrag[2];
            #pragma unroll
            for (int i = 0; i < 2; ++i) {
                const _Float16* ap = &Ab[(waveM * 32 + i * 16 + l15) * LDA + kk + aoff];
                v8h lo = *reinterpret_cast<const v8h*>(ap);        // K = kk+aoff+0..7
                v8h hi = *reinterpret_cast<const v8h*>(ap + 16);   // K = kk+aoff+16..23
                afrag[i] = __builtin_shufflevector(lo, hi,
                    0,1,2,3,4,5,6,7,8,9,10,11,12,13,14,15);
            }
            #pragma unroll
            for (int j = 0; j < 4; ++j) {
                const _Float16* bp = &Bb[(waveN * 64 + j * 16 + l15) * LDB + kk + boff];
                v8h blo = *reinterpret_cast<const v8h*>(bp);       // K = kk+boff+0..7
                v8h bhi = *reinterpret_cast<const v8h*>(bp + 16);  // K = kk+boff+16..23
                v16h bfrag = __builtin_shufflevector(blo, bhi,
                    0,1,2,3,4,5,6,7,8,9,10,11,12,13,14,15);
                #pragma unroll
                for (int i = 0; i < 2; ++i) {
                    acc[i][j] = __builtin_amdgcn_wmma_f32_16x16x32_f16(
                        /*neg_a=*/false, afrag[i],
                        /*neg_b=*/false, bfrag,
                        /*c_mod=*/(short)0, acc[i][j],
                        /*reuse_a=*/false, /*reuse_b=*/false);
                }
            }
        }

        // ---- stage next chunk into the other buffer, one barrier/iter ----
        if (hasNext) {
            const int nxt = cur ^ 1;
            #pragma unroll
            for (int i = 0; i < 8; ++i) {
                v4h h = { (_Float16)xr[i].x, (_Float16)xr[i].y,
                          (_Float16)xr[i].z, (_Float16)xr[i].w };
                *reinterpret_cast<v4h*>(&As[nxt][(i * 16 + xr0) * LDA + xc4 * 4]) = h;
            }
            #pragma unroll
            for (int i = 0; i < 4; ++i) {
                const int pr = prb + 2 * i;
                v8h wq;
                #pragma unroll
                for (int j = 0; j < 8; ++j)
                    wq[j] = (_Float16)(scale * (float)((wr[i] >> (4 * j)) & 0xF) + mzs);
                *reinterpret_cast<v8h*>(&Bs[nxt][wn * LDB + pr * 8]) = wq;
            }
            __syncthreads();
        }
    }

    // ---- epilogue: f32 C/D layout (VGPR r -> row r or r+8) ----
    const int m0 = bm0 + waveM * 32 + ((lane & 16) ? 8 : 0);
    const int n0 = bn0 + waveN * 64 + l15;
    #pragma unroll
    for (int i = 0; i < 2; ++i) {
        #pragma unroll
        for (int j = 0; j < 4; ++j) {
            float* op = Out + (size_t)(m0 + i * 16) * N + n0 + j * 16;
            #pragma unroll
            for (int r = 0; r < 8; ++r)
                op[(size_t)r * N] = acc[i][j][r];
        }
    }
}

extern "C" void kernel_launch(void* const* d_in, const int* in_sizes, int n_in,
                              void* d_out, int out_size, void* d_ws, size_t ws_size,
                              hipStream_t stream) {
    (void)n_in; (void)out_size; (void)d_ws; (void)ws_size;
    const float* X  = (const float*)d_in[0];
    const int*   QW = (const int*)d_in[1];
    const int*   QZ = (const int*)d_in[2];
    const float* SC = (const float*)d_in[3];
    float* Out = (float*)d_out;

    const int K = 4096;
    const int G = K / 128;             // 32 groups
    const int N = in_sizes[3] / G;     // scales is [G, N] -> 11008
    const int M = in_sizes[0] / K;     // x is [B*S, K]    -> 8192

    dim3 grid((unsigned)(N / BN), (unsigned)(M / BM));
    gptq_w4a16_wmma<<<grid, NTHREADS, 0, stream>>>(X, QW, QZ, SC, Out, M, N, K);
}
